// ResidualBlock_77764677861411
// MI455X (gfx1250) — matrix operand order repaired and checked by reading, not compiled, not run on hardware
//
#include <hip/hip_runtime.h>

// ---------------------------------------------------------------------------
// BitNet ternary dilated-conv residual block on gfx1250 (MI455X).
// Convs are GEMMs fed by v_wmma_f32_16x16x32_bf16. The bf16 intermediate is
// stored transposed [B,L,C] so conv-B stages activations into LDS with
// GLOBAL_LOAD_ASYNC_TO_LDS_B128 (ASYNCcnt) and conv-A's epilogue is a single
// b128 store per fragment. LDS is double-buffered to overlap staging w/ WMMA.
// ---------------------------------------------------------------------------

typedef __attribute__((ext_vector_type(16))) __bf16    v16bf;
typedef __attribute__((ext_vector_type(8)))  float     v8f;
typedef __attribute__((ext_vector_type(4)))  unsigned  v4u;

#define C_IN      512
#define L_LEN     8192
#define BATCH     8
#define KTAPS     3
#define W_ELEMS   (C_IN * C_IN * KTAPS)          // 786432 per tensor
#define O_TILE    64                              // per block
#define L_TILE    128                             // per block
#define KC        32                              // channels per K step
#define NKC       (C_IN / KC)                     // 16 K steps
#define LDS_CSTR  40                              // ushorts per LDS column (80B, 16B aligned)
#define PW_STRIDE (KTAPS * 16 * 32 * 32 * 8)      // dwords per packed tensor = 393216

union ABfrag { v4u u[2]; v16bf v; };

static __device__ inline unsigned short f2bf(float f) {
    unsigned u = __builtin_bit_cast(unsigned, f);
    u += 0x7FFFu + ((u >> 16) & 1u);              // round-to-nearest-even
    return (unsigned short)(u >> 16);
}

// Async global->LDS copy, 16B per lane, tracked on ASYNCcnt (CDNA5).
static __device__ inline void async_copy_b128(unsigned ldsOff, const void* gaddr) {
    asm volatile("global_load_async_to_lds_b128 %0, %1, off"
                 :: "v"(ldsOff), "v"(gaddr) : "memory");
}
static __device__ inline void async_fence() {
    asm volatile("s_wait_asynccnt 0x0" ::: "memory");
}
// Generic pointer to an LDS object: low 32 bits are the LDS byte offset.
static __device__ inline unsigned lds_off(const void* p) {
    return (unsigned)(unsigned long long)p;
}

// -------------------------- scale accumulators -----------------------------
__global__ void zero_accum_kernel(float* accum) {
    if (threadIdx.x < 6) accum[threadIdx.x] = 0.0f;
}

__global__ __launch_bounds__(256) void absmean_kernel(
    const float* __restrict__ w0, const float* __restrict__ w1,
    const float* __restrict__ w2, const float* __restrict__ w3,
    const float* __restrict__ w4, const float* __restrict__ w5,
    float* __restrict__ accum) {
    const int ti    = blockIdx.x / 96;
    const int chunk = blockIdx.x % 96;
    const float* w;
    switch (ti) {
        case 0: w = w0; break; case 1: w = w1; break; case 2: w = w2; break;
        case 3: w = w3; break; case 4: w = w4; break; default: w = w5; break;
    }
    float s = 0.0f;
    const int base = chunk * 8192 + threadIdx.x;
    #pragma unroll 4
    for (int i = 0; i < 32; ++i) s += fabsf(w[base + i * 256]);
    __shared__ float red[256];
    red[threadIdx.x] = s;
    __syncthreads();
    for (int off = 128; off > 0; off >>= 1) {
        if (threadIdx.x < off) red[threadIdx.x] += red[threadIdx.x + off];
        __syncthreads();
    }
    if (threadIdx.x == 0) atomicAdd(&accum[ti], red[0]);
}

// --------------- ternary quantize + pack into WMMA A-fragment layout -------
// A-matrix 16x32 bf16 lane layout (ISA 7.12.2):
//   lanes 0-15 : M = lane,    VGPR v<4 -> K = 2v,2v+1 ; v>=4 -> K = 2v+8,2v+9
//   lanes 16-31: M = lane-16, VGPR v<4 -> K = 2v+8    ; v>=4 -> K = 2v+16
__global__ __launch_bounds__(256) void pack_kernel(
    const float* __restrict__ w0, const float* __restrict__ w1,
    const float* __restrict__ w2, const float* __restrict__ w3,
    const float* __restrict__ w4, const float* __restrict__ w5,
    const float* __restrict__ accum, unsigned* __restrict__ pw) {
    const int lane  = threadIdx.x & 31;
    const int wave  = threadIdx.x >> 5;
    const int gtile = blockIdx.x * 8 + wave;      // 6*3*16*32 = 9216 tiles
    if (gtile >= 9216) return;
    const int ti  = gtile / 1536;
    int r         = gtile - ti * 1536;
    const int tap = r >> 9;
    r &= 511;
    const int kt  = r >> 5;
    const int ot  = r & 31;

    const float* w;
    switch (ti) {
        case 0: w = w0; break; case 1: w = w1; break; case 2: w = w2; break;
        case 3: w = w3; break; case 4: w = w4; break; default: w = w5; break;
    }
    const float scale = accum[ti] * (1.0f / (float)W_ELEMS) + 1e-5f;
    const float inv   = 1.0f / scale;

    const int m  = lane & 15;
    const int kl = lane >> 4;
    const int o  = ot * 16 + m;
    unsigned* dst = pw + (size_t)ti * PW_STRIDE
                  + (((size_t)(tap * 16 + kt) * 32 + ot) * 32 + lane) * 8;
    #pragma unroll
    for (int v = 0; v < 8; ++v) {
        const int k0 = (v < 4) ? (2 * v + 8 * kl) : (2 * v + 8 + 8 * kl);
        const int i0 = kt * 32 + k0;
        unsigned bits = 0;
        #pragma unroll
        for (int h = 0; h < 2; ++h) {
            const float wv = w[((size_t)o * C_IN + (i0 + h)) * KTAPS + tap];
            const float q  = rintf(wv * inv);
            unsigned hb = 0;
            if (q > 0.5f)       hb = 0x3F80u;      // +1 in bf16
            else if (q < -0.5f) hb = 0xBF80u;      // -1 in bf16
            bits |= hb << (16 * h);
        }
        dst[v] = bits;
    }
}

// ------------------------------ LDS staging --------------------------------
// conv A: input f32 in [B,C,L]; load coalesced along L, cvt, ds_store_b16.
template <int D, int COLS>
static __device__ inline void stage_tile(unsigned short* buf, const float* in,
                                         int b, int l0, int kc) {
    for (int idx = threadIdx.x; idx < KC * COLS; idx += 256) {
        const int ch  = idx / COLS;
        const int col = idx - ch * COLS;
        const int l   = l0 - D + col;
        unsigned short v = 0;
        if (l >= 0 && l < L_LEN)
            v = f2bf(in[((size_t)(b * C_IN + kc * KC + ch)) * L_LEN + l]);
        buf[col * LDS_CSTR + ch] = v;
    }
}
// conv B: input bf16 in transposed [B,L,C]; 16B-contiguous rows -> async b128.
template <int D, int COLS>
static __device__ inline void stage_tile(unsigned short* buf,
                                         const unsigned short* in,
                                         int b, int l0, int kc) {
    for (int it = threadIdx.x; it < COLS * 4; it += 256) {
        const int col = it >> 2;
        const int sub = it & 3;                    // 4 x b128 = 32 channels
        const int l   = l0 - D + col;
        unsigned short* dst = &buf[col * LDS_CSTR + sub * 8];
        if (l >= 0 && l < L_LEN) {
            async_copy_b128(lds_off(dst),
                in + ((size_t)(b * L_LEN + l)) * C_IN + kc * KC + sub * 8);
        } else {
            v4u z = {0u, 0u, 0u, 0u};
            *(v4u*)dst = z;                        // ds_store_b128 zero halo
        }
    }
}

// ------------------------------ epilogues ----------------------------------
// conv A: scale+bias+leaky, pack 8 bf16, single b128 store into [B,L,C].
static __device__ inline void epilogue_store(unsigned short* out, const float*,
                                             int b, int n, int oB, int hi,
                                             v8f a, float scale,
                                             const float* bias_r) {
    unsigned d[4];
    #pragma unroll
    for (int p = 0; p < 4; ++p) {
        float v0 = a[2 * p]     * scale + bias_r[2 * p];
        float v1 = a[2 * p + 1] * scale + bias_r[2 * p + 1];
        v0 = (v0 >= 0.0f) ? v0 : 0.1f * v0;
        v1 = (v1 >= 0.0f) ? v1 : 0.1f * v1;
        d[p] = (unsigned)f2bf(v0) | ((unsigned)f2bf(v1) << 16);
    }
    v4u val = {d[0], d[1], d[2], d[3]};
    *(v4u*)&out[((size_t)(b * L_LEN + n)) * C_IN + oB + hi * 8] = val;
}
// conv B: scale+bias+residual, f32 store into standard [B,C,L].
static __device__ inline void epilogue_store(float* out, const float* res,
                                             int b, int n, int oB, int hi,
                                             v8f a, float scale,
                                             const float* bias_r) {
    #pragma unroll
    for (int rr = 0; rr < 8; ++rr) {
        const int o = oB + rr + hi * 8;
        const size_t g = ((size_t)(b * C_IN + o)) * L_LEN + n;
        out[g] = a[rr] * scale + bias_r[rr] + res[g];
    }
}

// ------------------------------- conv kernel -------------------------------
// y[b,o,l] = scale * sum_{i,tap} t[o,i,tap] * x[b, i, l + (tap-1)*D] + bias[o]
// Block: 256 threads = 8 waves -> tile 64 O x 128 L. Wave (ow,nw): 16 O x 64 L.
// Double-buffered LDS: stage K-step kc+1 while WMMAs consume kc.
template <int D, typename TIN, typename TOUT>
__global__ __launch_bounds__(256) void bitconv_kernel(
    const TIN* __restrict__ in, const float* __restrict__ bias,
    const float* __restrict__ scaleSum, const unsigned* __restrict__ pw,
    const float* __restrict__ res, TOUT* __restrict__ out) {

    constexpr int COLS = L_TILE + 2 * D;           // dilation halo both sides
    __shared__ unsigned short lds[2][COLS * LDS_CSTR];

    const int lane = threadIdx.x & 31;
    const int wave = threadIdx.x >> 5;
    const int ow   = wave & 3;                     // o sub-tile (16 rows each)
    const int nw   = wave >> 2;                    // l half (64 cols each)
    const int b    = blockIdx.z;
    const int l0   = blockIdx.x * L_TILE;
    const int oB   = blockIdx.y * O_TILE + ow * 16;
    const int m    = lane & 15;
    const int hi   = lane >> 4;
    const int otG  = blockIdx.y * 4 + ow;

    v8f acc[4] = {};

    stage_tile<D, COLS>(lds[0], in, b, l0, 0);     // prologue stage

    for (int kc = 0; kc < NKC; ++kc) {
        async_fence();                             // own async copies done
        __syncthreads();                           // buffer kc ready, prev reads done
        if (kc + 1 < NKC)
            stage_tile<D, COLS>(lds[(kc + 1) & 1], in, b, l0, kc + 1);
        const unsigned short* buf = lds[kc & 1];

        if (kc + 1 < NKC)                          // warm L2/WGP$ for next A frag
            __builtin_prefetch(pw + (((size_t)(kc + 1) * 32 + otG) * 32 + lane) * 8, 0, 0);

        #pragma unroll
        for (int tap = 0; tap < KTAPS; ++tap) {
            // A fragment: prepacked ternary bf16 weights, 2x b128 per lane
            const v4u* ap = (const v4u*)(pw
                + (((size_t)(tap * 16 + kc) * 32 + otG) * 32 + lane) * 8);
            ABfrag a;
            a.u[0] = ap[0];
            a.u[1] = ap[1];
            #pragma unroll
            for (int j = 0; j < 4; ++j) {
                // B fragment 32x16: lanes 0-15 K=0..7,16..23, lanes 16-31 K=8..15,24..31
                const int col = nw * 64 + j * 16 + m + tap * D;
                const v4u* bp = (const v4u*)(&buf[col * LDS_CSTR + hi * 8]);
                ABfrag bf;
                bf.u[0] = bp[0];
                bf.u[1] = bp[2];
                acc[j] = __builtin_amdgcn_wmma_f32_16x16x32_bf16(
                    false, a.v, false, bf.v, (short)0, acc[j], false, false);
            }
        }
    }

    // Epilogue: scale, bias, activation/residual, store.
    const float scale = scaleSum[0] * (1.0f / (float)W_ELEMS) + 1e-5f;
    float bias_r[8];
    #pragma unroll
    for (int rr = 0; rr < 8; ++rr) bias_r[rr] = bias[oB + rr + hi * 8];

    #pragma unroll
    for (int j = 0; j < 4; ++j) {
        const int n = l0 + nw * 64 + j * 16 + m;
        epilogue_store(out, res, b, n, oB, hi, acc[j], scale, bias_r);
    }
}

// ------------------------------- launcher ----------------------------------
extern "C" void kernel_launch(void* const* d_in, const int* in_sizes, int n_in,
                              void* d_out, int out_size, void* d_ws, size_t ws_size,
                              hipStream_t stream) {
    (void)in_sizes; (void)n_in; (void)out_size; (void)ws_size;
    const float* x = (const float*)d_in[0];
    const float* w[6];
    const float* bias[6];
    for (int ti = 0; ti < 6; ++ti) {
        w[ti]    = (const float*)d_in[1 + 2 * ti];
        bias[ti] = (const float*)d_in[2 + 2 * ti];
    }
    float*          accum = (float*)d_ws;
    unsigned*       pw    = (unsigned*)((char*)d_ws + 256);
    unsigned short* bufH  = (unsigned short*)((char*)d_ws + 256 + (size_t)6 * PW_STRIDE * 4);
    float*          out   = (float*)d_out;

    zero_accum_kernel<<<1, 32, 0, stream>>>(accum);
    absmean_kernel<<<6 * 96, 256, 0, stream>>>(w[0], w[1], w[2], w[3], w[4], w[5], accum);
    pack_kernel<<<9216 / 8, 256, 0, stream>>>(w[0], w[1], w[2], w[3], w[4], w[5], accum, pw);

    dim3 grid(L_LEN / L_TILE, C_IN / O_TILE, BATCH);   // 64 x 8 x 8
    dim3 blk(256);

    // Branch 0 (d=1): h = leaky(convA(x)) -> bufH[B,L,C] ; out = x + convB(h)
    bitconv_kernel<1, float, unsigned short><<<grid, blk, 0, stream>>>(
        x, bias[0], accum + 0, pw + (size_t)0 * PW_STRIDE, nullptr, bufH);
    bitconv_kernel<1, unsigned short, float><<<grid, blk, 0, stream>>>(
        bufH, bias[1], accum + 1, pw + (size_t)1 * PW_STRIDE, x, out);

    // Branch 1 (d=3)
    bitconv_kernel<3, float, unsigned short><<<grid, blk, 0, stream>>>(
        out, bias[2], accum + 2, pw + (size_t)2 * PW_STRIDE, nullptr, bufH);
    bitconv_kernel<1, unsigned short, float><<<grid, blk, 0, stream>>>(
        bufH, bias[3], accum + 3, pw + (size_t)3 * PW_STRIDE, out, out);

    // Branch 2 (d=5)
    bitconv_kernel<5, float, unsigned short><<<grid, blk, 0, stream>>>(
        out, bias[4], accum + 4, pw + (size_t)4 * PW_STRIDE, nullptr, bufH);
    bitconv_kernel<1, unsigned short, float><<<grid, blk, 0, stream>>>(
        bufH, bias[5], accum + 5, pw + (size_t)5 * PW_STRIDE, out, out);
}
